// GKNet_63857573757420
// MI455X (gfx1250) — matrix operand order repaired and checked by reading, not compiled, not run on hardware
//
#include <hip/hip_runtime.h>

// ---------------------------------------------------------------------------
// GKNet forward for MI455X (gfx1250).
//  - Aggregation matmuls (dominant: 6 x [2048x2048]@[2048x288] ~ 14.5 GFLOP)
//    run on v_wmma_f32_16x16x32_bf16.
//  - Layer 1: PNA weights (binary-mean / dist-mean / softmax / softmin) are
//    precomputed ONCE into bf16 fragments pre-swizzled to the A-operand VGPR
//    layout; the WMMA loop streams them from L2. Each wave covers TWO column
//    tiles (12 WMMAs / K-step) to halve the dominant weight-fragment traffic.
//  - Layer 0 (single column tile) keeps fused on-the-fly weight generation.
//  - Everything else is cheap scalar f32.
// ---------------------------------------------------------------------------

constexpr int   kN         = 2048;
constexpr int   kP         = 9;
constexpr int   kHid       = 32;
constexpr float kAvgLogDeg = 7.625f;
constexpr float kEps       = 1e-8f;
constexpr int   kC1        = 16;            // layer-0 padded cols (9 -> 16)
constexpr int   kC2        = kP * kHid;     // layer-1 cols = 288
constexpr int   kTB        = kN / 16;       // 128 row tiles
constexpr int   kKT        = kN / 32;       // 64 K tiles
constexpr int   kFragElems = 32 * 16;       // elements per (tile,ktile) fragment
// elements per weight-variant plane: 128*64*512 = 4,194,304 (8 MB bf16)
constexpr size_t kWPlane   = (size_t)kTB * kKT * kFragElems;

typedef __attribute__((ext_vector_type(16))) __bf16 v16bf;
typedef __attribute__((ext_vector_type(8)))  float  v8f;

// ---------------------------- block reductions -----------------------------
__device__ inline float blockReduceSum(float v, float* red) {
  int t = threadIdx.x;
  red[t] = v;
  __syncthreads();
  for (int s = 128; s > 0; s >>= 1) {
    if (t < s) red[t] += red[t + s];
    __syncthreads();
  }
  float r = red[0];
  __syncthreads();
  return r;
}

__device__ inline float blockReduceMax(float v, float* red) {
  int t = threadIdx.x;
  red[t] = v;
  __syncthreads();
  for (int s = 128; s > 0; s >>= 1) {
    if (t < s) red[t] = fmaxf(red[t], red[t + s]);
    __syncthreads();
  }
  float r = red[0];
  __syncthreads();
  return r;
}

// ------------------------- per-row adjacency stats -------------------------
// stats[row][0..7] = { 1/degb, 1/(rowsum+eps), amax, 1/sum(exp(a-amax)),
//                      nmax, 1/sum(exp(-a-nmax)), s, 1/s }
__global__ __launch_bounds__(256)
void rowstats_kernel(const float* __restrict__ A, float* __restrict__ stats) {
  __shared__ float red[256];
  const int row = blockIdx.x;
  const int t   = threadIdx.x;
  const float* a = A + (size_t)row * kN;

  float cnt = 0.f, sum = 0.f, amax = -1e30f, nmax = -1e30f;
  for (int m = t; m < kN; m += 256) {
    float v = a[m];
    sum += v;
    if (v > 0.f) {
      cnt += 1.f;
      amax = fmaxf(amax, v);
      nmax = fmaxf(nmax, -v);
    }
  }
  cnt  = blockReduceSum(cnt, red);
  sum  = blockReduceSum(sum, red);
  amax = blockReduceMax(amax, red);
  nmax = blockReduceMax(nmax, red);
  if (cnt == 0.f) { amax = 0.f; nmax = 0.f; }

  float sexp = 0.f, snexp = 0.f;
  for (int m = t; m < kN; m += 256) {
    float v = a[m];
    if (v > 0.f) {
      sexp  += __expf(v - amax);
      snexp += __expf(-v - nmax);
    }
  }
  sexp  = blockReduceSum(sexp, red);
  snexp = blockReduceSum(snexp, red);

  if (t == 0) {
    float degb = fmaxf(cnt, 1.f);
    float s = __logf(degb + 1.f) / kAvgLogDeg;
    float* st = stats + row * 8;
    st[0] = 1.f / degb;
    st[1] = 1.f / (sum + kEps);
    st[2] = amax;
    st[3] = sexp > 0.f ? 1.f / sexp : 0.f;
    st[4] = nmax;
    st[5] = snexp > 0.f ? 1.f / snexp : 0.f;
    st[6] = s;
    st[7] = 1.f / s;
  }
}

// ------------- on-the-fly weight fragment generation (A-operand) -----------
// A-fragment k order used here (contiguous, B matches): VGPR i holds K pair at 16*hi+2*i.
__device__ inline void make_weight_frags(const float* __restrict__ arow, int kb, int hi,
                                         float invdeg, float invsum, float amax,
                                         float invsexp, float nmax, float invsnexp,
                                         v16bf& wbv, v16bf& wav, v16bf& spv, v16bf& smv) {
#pragma unroll
  for (int i = 0; i < 8; ++i) {
    const int k0 = kb + 16 * hi + 2 * i;
    const float a0 = arow[k0];
    const float a1 = arow[k0 + 1];
    const float m0 = a0 > 0.f ? 1.f : 0.f;
    const float m1 = a1 > 0.f ? 1.f : 0.f;
    wbv[2 * i]     = (__bf16)(m0 * invdeg);
    wbv[2 * i + 1] = (__bf16)(m1 * invdeg);
    wav[2 * i]     = (__bf16)(a0 * invsum);
    wav[2 * i + 1] = (__bf16)(a1 * invsum);
    spv[2 * i]     = (__bf16)(m0 * __expf(a0 - amax) * invsexp);
    spv[2 * i + 1] = (__bf16)(m1 * __expf(a1 - amax) * invsexp);
    smv[2 * i]     = (__bf16)(m0 * __expf(-a0 - nmax) * invsnexp);
    smv[2 * i + 1] = (__bf16)(m1 * __expf(-a1 - nmax) * invsnexp);
  }
}

// ------------------ prep: H (f32, row-major) -> bf16 H^T, (H^2)^T ----------
__global__ void prep0_kernel(const float* __restrict__ X,
                             __bf16* __restrict__ HbT, __bf16* __restrict__ H2bT) {
  int idx = blockIdx.x * blockDim.x + threadIdx.x;   // over kC1*kN
  if (idx >= kC1 * kN) return;
  int c = idx / kN, m = idx % kN;
  float v = (c < kP) ? X[(size_t)m * kP + c] : 0.f;
  HbT[idx]  = (__bf16)v;
  H2bT[idx] = (__bf16)(v * v);
}

__global__ void prep1_kernel(const float* __restrict__ H,
                             __bf16* __restrict__ HbT, __bf16* __restrict__ H2bT) {
  int idx = blockIdx.x * blockDim.x + threadIdx.x;   // over kC2*kN
  if (idx >= kC2 * kN) return;
  int c = idx / kN, m = idx % kN;
  float v = H[(size_t)m * kC2 + c];
  HbT[idx]  = (__bf16)v;
  H2bT[idx] = (__bf16)(v * v);
}

// ---------- precompute swizzled bf16 weight fragments (layer 1) ------------
// Wbuf[variant][tb][kt][lane][16], variants {wb, wa, sp, sm}; one thread per
// (tb, kt, lane) writes its 16-element fragment for all 4 variants.
__global__ __launch_bounds__(256)
void wprep_kernel(const float* __restrict__ A, const float* __restrict__ stats,
                  __bf16* __restrict__ Wbuf) {
  int idx = blockIdx.x * blockDim.x + threadIdx.x;   // kTB*kKT*32 = 262144
  if (idx >= kTB * kKT * 32) return;
  const int lane = idx & 31;
  const int kt   = (idx >> 5) & (kKT - 1);
  const int tb   = idx >> 11;
  const int lrow = lane & 15;
  const int hi   = lane >> 4;

  const int row = tb * 16 + lrow;
  const float* st = stats + row * 8;
  const float* arow = A + (size_t)row * kN;

  v16bf wbv = {}, wav = {}, spv = {}, smv = {};
  make_weight_frags(arow, kt * 32, hi, st[0], st[1], st[2], st[3], st[4], st[5],
                    wbv, wav, spv, smv);

  const size_t frag = ((size_t)tb * kKT + kt) * kFragElems + lane * 16;
  *(v16bf*)(Wbuf + 0 * kWPlane + frag) = wbv;
  *(v16bf*)(Wbuf + 1 * kWPlane + frag) = wav;
  *(v16bf*)(Wbuf + 2 * kWPlane + frag) = spv;
  *(v16bf*)(Wbuf + 3 * kWPlane + frag) = smv;
}

// --------------------- shared WMMA epilogue (7 features) -------------------
__device__ inline void write_features(float* __restrict__ F, int Cp, int n0, int c0,
                                      int hi, int lrow,
                                      const v8f& acc_sm, const v8f& acc_sp,
                                      const v8f& acc_mb, const v8f& acc_vb,
                                      const v8f& acc_ma, const v8f& acc_va) {
  const size_t plane = (size_t)kN * Cp;
#pragma unroll
  for (int r = 0; r < 8; ++r) {
    const int orow = n0 + r + 8 * hi;       // C/D layout: M = r (+8 for hi lanes)
    const int ocol = c0 + lrow;
    const size_t idx = (size_t)orow * Cp + ocol;
    const float sm = acc_sm[r], sp = acc_sp[r];
    const float mb = acc_mb[r];
    const float vb = fmaxf(acc_vb[r] - mb * mb, 0.f);
    const float ma = acc_ma[r];
    const float va = fmaxf(acc_va[r] - ma * ma, 0.f);
    F[0 * plane + idx] = sm;
    F[1 * plane + idx] = sp;
    F[2 * plane + idx] = mb;
    F[3 * plane + idx] = sqrtf(vb + kEps);
    F[4 * plane + idx] = vb;
    F[5 * plane + idx] = ma;
    F[6 * plane + idx] = sqrtf(va + kEps);
  }
}

// ------------- layer-0 aggregation: fused weight gen + WMMA ----------------
__global__ __launch_bounds__(32)
void agg_wmma_fused_kernel(const float* __restrict__ A, const float* __restrict__ stats,
                           const __bf16* __restrict__ HbT, const __bf16* __restrict__ H2bT,
                           float* __restrict__ F, int Cp) {
  const int n0   = blockIdx.x * 16;
  const int c0   = blockIdx.y * 16;
  const int lane = threadIdx.x;
  const int lrow = lane & 15;
  const int hi   = lane >> 4;

  const int row = n0 + lrow;
  const float* st = stats + row * 8;
  const float invdeg = st[0], invsum = st[1];
  const float amax = st[2], invsexp = st[3];
  const float nmax = st[4], invsnexp = st[5];

  const int colc = c0 + lrow;
  const __bf16* hbBase  = HbT  + (size_t)colc * kN + 16 * hi;
  const __bf16* h2bBase = H2bT + (size_t)colc * kN + 16 * hi;
  const float*  arow    = A + (size_t)row * kN;

  v8f acc_sm = {}, acc_sp = {}, acc_mb = {}, acc_vb = {}, acc_ma = {}, acc_va = {};

  for (int kb = 0; kb < kN; kb += 32) {
    v16bf wbv = {}, wav = {}, spv = {}, smv = {};
    make_weight_frags(arow, kb, hi, invdeg, invsum, amax, invsexp, nmax, invsnexp,
                      wbv, wav, spv, smv);
    const v16bf hb  = *(const v16bf*)(hbBase + kb);
    const v16bf h2b = *(const v16bf*)(h2bBase + kb);

    acc_sm = __builtin_amdgcn_wmma_f32_16x16x32_bf16(false, smv, false, hb,  (short)0, acc_sm, false, false);
    acc_sp = __builtin_amdgcn_wmma_f32_16x16x32_bf16(false, spv, false, hb,  (short)0, acc_sp, false, false);
    acc_mb = __builtin_amdgcn_wmma_f32_16x16x32_bf16(false, wbv, false, hb,  (short)0, acc_mb, false, false);
    acc_vb = __builtin_amdgcn_wmma_f32_16x16x32_bf16(false, wbv, false, h2b, (short)0, acc_vb, false, false);
    acc_ma = __builtin_amdgcn_wmma_f32_16x16x32_bf16(false, wav, false, hb,  (short)0, acc_ma, false, false);
    acc_va = __builtin_amdgcn_wmma_f32_16x16x32_bf16(false, wav, false, h2b, (short)0, acc_va, false, false);
  }

  write_features(F, Cp, n0, c0, hi, lrow, acc_sm, acc_sp, acc_mb, acc_vb, acc_ma, acc_va);
}

// ---------- layer-1 aggregation: stream precomputed W, 2 column tiles ------
__global__ __launch_bounds__(32)
void agg_wmma_pre_kernel(const __bf16* __restrict__ Wbuf,
                         const __bf16* __restrict__ HbT, const __bf16* __restrict__ H2bT,
                         float* __restrict__ F, int Cp) {
  const int tb   = blockIdx.x;
  const int n0   = tb * 16;
  const int c0   = blockIdx.y * 32;        // two 16-wide column tiles
  const int lane = threadIdx.x;
  const int lrow = lane & 15;
  const int hi   = lane >> 4;

  const __bf16* hbBase0  = HbT  + (size_t)(c0 + lrow) * kN + 16 * hi;
  const __bf16* h2bBase0 = H2bT + (size_t)(c0 + lrow) * kN + 16 * hi;
  const __bf16* hbBase1  = hbBase0  + (size_t)16 * kN;
  const __bf16* h2bBase1 = h2bBase0 + (size_t)16 * kN;
  const __bf16* wBase    = Wbuf + (size_t)tb * kKT * kFragElems + lane * 16;

  v8f a_sm0 = {}, a_sp0 = {}, a_mb0 = {}, a_vb0 = {}, a_ma0 = {}, a_va0 = {};
  v8f a_sm1 = {}, a_sp1 = {}, a_mb1 = {}, a_vb1 = {}, a_ma1 = {}, a_va1 = {};

  for (int kt = 0; kt < kKT; ++kt) {
    const __bf16* wf = wBase + (size_t)kt * kFragElems;
    const v16bf wbv = *(const v16bf*)(wf);
    const v16bf wav = *(const v16bf*)(wf + kWPlane);
    const v16bf spv = *(const v16bf*)(wf + 2 * kWPlane);
    const v16bf smv = *(const v16bf*)(wf + 3 * kWPlane);
    const v16bf hb0  = *(const v16bf*)(hbBase0  + kt * 32);
    const v16bf h2b0 = *(const v16bf*)(h2bBase0 + kt * 32);
    const v16bf hb1  = *(const v16bf*)(hbBase1  + kt * 32);
    const v16bf h2b1 = *(const v16bf*)(h2bBase1 + kt * 32);

    a_sm0 = __builtin_amdgcn_wmma_f32_16x16x32_bf16(false, smv, false, hb0,  (short)0, a_sm0, false, false);
    a_sp0 = __builtin_amdgcn_wmma_f32_16x16x32_bf16(false, spv, false, hb0,  (short)0, a_sp0, false, false);
    a_mb0 = __builtin_amdgcn_wmma_f32_16x16x32_bf16(false, wbv, false, hb0,  (short)0, a_mb0, false, false);
    a_vb0 = __builtin_amdgcn_wmma_f32_16x16x32_bf16(false, wbv, false, h2b0, (short)0, a_vb0, false, false);
    a_ma0 = __builtin_amdgcn_wmma_f32_16x16x32_bf16(false, wav, false, hb0,  (short)0, a_ma0, false, false);
    a_va0 = __builtin_amdgcn_wmma_f32_16x16x32_bf16(false, wav, false, h2b0, (short)0, a_va0, false, false);

    a_sm1 = __builtin_amdgcn_wmma_f32_16x16x32_bf16(false, smv, false, hb1,  (short)0, a_sm1, false, false);
    a_sp1 = __builtin_amdgcn_wmma_f32_16x16x32_bf16(false, spv, false, hb1,  (short)0, a_sp1, false, false);
    a_mb1 = __builtin_amdgcn_wmma_f32_16x16x32_bf16(false, wbv, false, hb1,  (short)0, a_mb1, false, false);
    a_vb1 = __builtin_amdgcn_wmma_f32_16x16x32_bf16(false, wbv, false, h2b1, (short)0, a_vb1, false, false);
    a_ma1 = __builtin_amdgcn_wmma_f32_16x16x32_bf16(false, wav, false, hb1,  (short)0, a_ma1, false, false);
    a_va1 = __builtin_amdgcn_wmma_f32_16x16x32_bf16(false, wav, false, h2b1, (short)0, a_va1, false, false);
  }

  write_features(F, Cp, n0, c0,      hi, lrow, a_sm0, a_sp0, a_mb0, a_vb0, a_ma0, a_va0);
  write_features(F, Cp, n0, c0 + 16, hi, lrow, a_sm1, a_sp1, a_mb1, a_vb1, a_ma1, a_va1);
}

// ------------------- theta0 projection + bias + leaky relu -----------------
__global__ void theta0_kernel(const float* __restrict__ F0, const float* __restrict__ th,
                              const float* __restrict__ bias, float* __restrict__ H1) {
  int idx = blockIdx.x * blockDim.x + threadIdx.x;   // N*P*HID
  if (idx >= kN * kP * kHid) return;
  const int o = idx % kHid;
  const int p = (idx / kHid) % kP;
  const int n = idx / (kHid * kP);
  float acc = bias[o];
#pragma unroll
  for (int f = 0; f < 7; ++f)
    acc += F0[((size_t)f * kN + n) * kC1 + p] * th[f * kHid + o];
  H1[(size_t)n * kC2 + p * kHid + o] = acc > 0.f ? acc : 0.01f * acc;
}

// -------------------- temporal conv0 (1x3 SAME) + GLU ----------------------
__global__ void tconv0_glu_kernel(const float* __restrict__ H1, const float* __restrict__ w,
                                  const float* __restrict__ b, float* __restrict__ H2) {
  int idx = blockIdx.x * blockDim.x + threadIdx.x;   // N*P*HID
  if (idx >= kN * kP * kHid) return;
  const int oc = idx % kHid;
  const int p  = (idx / kHid) % kP;
  const int n  = idx / (kHid * kP);
  float acc0 = b[oc];
  float acc1 = b[oc + kHid];
#pragma unroll
  for (int dt = 0; dt < 3; ++dt) {
    const int pp = p + dt - 1;
    if (pp < 0 || pp >= kP) continue;
    const float* hrow = H1 + (size_t)n * kC2 + pp * kHid;
    const float* w0 = w + ((size_t)oc * kHid) * 3 + dt;            // [64][32][1][3]
    const float* w1 = w + ((size_t)(oc + kHid) * kHid) * 3 + dt;
    for (int ic = 0; ic < kHid; ++ic) {
      acc0 += hrow[ic] * w0[ic * 3];
      acc1 += hrow[ic] * w1[ic * 3];
    }
  }
  const float hv = H1[(size_t)n * kC2 + p * kHid + oc];
  H2[(size_t)n * kC2 + p * kHid + oc] = (acc0 + hv) * (1.f / (1.f + __expf(-acc1)));
}

// ---------- theta1 projection with PNA scalers + bias + leaky relu ---------
__global__ void theta1_kernel(const float* __restrict__ F1, const float* __restrict__ stats,
                              const float* __restrict__ th, const float* __restrict__ bias,
                              float* __restrict__ H3) {
  int idx = blockIdx.x * blockDim.x + threadIdx.x;   // N*P*HID
  if (idx >= kN * kP * kHid) return;
  const int o = idx % kHid;
  const int p = (idx / kHid) % kP;
  const int n = idx / (kHid * kP);
  const float s     = stats[n * 8 + 6];
  const float inv_s = stats[n * 8 + 7];
  float a0 = 0.f, a1 = 0.f, a2 = 0.f;
  for (int f = 0; f < 7; ++f) {
    const float* Fr = F1 + ((size_t)f * kN + n) * kC2 + p * kHid;
    for (int ch = 0; ch < kHid; ++ch) {
      const float v = Fr[ch];
      const int base = f * kHid + ch;
      a0 += v * th[(size_t)base * kHid + o];
      a1 += v * th[(size_t)(224 + base) * kHid + o];
      a2 += v * th[(size_t)(448 + base) * kHid + o];
    }
  }
  float acc = a0 + s * a1 + inv_s * a2 + bias[o];
  H3[(size_t)n * kC2 + p * kHid + o] = acc > 0.f ? acc : 0.01f * acc;
}

// ------------------ temporal conv1 (1x3 SAME) + sigmoid --------------------
__global__ void tconv1_sig_kernel(const float* __restrict__ H3, const float* __restrict__ w,
                                  const float* __restrict__ b, float* __restrict__ H4) {
  int idx = blockIdx.x * blockDim.x + threadIdx.x;   // N*P*HID
  if (idx >= kN * kP * kHid) return;
  const int oc = idx % kHid;
  const int p  = (idx / kHid) % kP;
  const int n  = idx / (kHid * kP);
  float acc = b[oc];
#pragma unroll
  for (int dt = 0; dt < 3; ++dt) {
    const int pp = p + dt - 1;
    if (pp < 0 || pp >= kP) continue;
    const float* hrow = H3 + (size_t)n * kC2 + pp * kHid;
    const float* w0 = w + ((size_t)oc * kHid) * 3 + dt;            // [32][32][1][3]
    for (int ic = 0; ic < kHid; ++ic)
      acc += hrow[ic] * w0[ic * 3];
  }
  const float hv = H3[(size_t)n * kC2 + p * kHid + oc];
  H4[(size_t)n * kC2 + p * kHid + oc] = 1.f / (1.f + __expf(-(acc + hv)));
}

// ---------------- head: 1x1 conv -> time linear -> ReLU --------------------
__global__ void out_kernel(const float* __restrict__ H4, const float* __restrict__ wc,
                           const float* __restrict__ bc, const float* __restrict__ lw,
                           const float* __restrict__ lb, float* __restrict__ out) {
  int n = blockIdx.x * blockDim.x + threadIdx.x;
  if (n >= kN) return;
  float y[kP];
#pragma unroll
  for (int p = 0; p < kP; ++p) {
    float a = bc[0];
    for (int c = 0; c < kHid; ++c)
      a += H4[(size_t)n * kC2 + p * kHid + c] * wc[c];
    y[p] = a;
  }
#pragma unroll
  for (int q = 0; q < kP; ++q) {
    float a = lb[q];
    for (int p = 0; p < kP; ++p)
      a += y[p] * lw[q * kP + p];
    out[(size_t)n * kP + q] = fmaxf(a, 0.f);
  }
}

// ---------------------------------------------------------------------------
extern "C" void kernel_launch(void* const* d_in, const int* in_sizes, int n_in,
                              void* d_out, int out_size, void* d_ws, size_t ws_size,
                              hipStream_t stream) {
  (void)in_sizes; (void)n_in; (void)out_size; (void)ws_size;

  const float* X        = (const float*)d_in[0];
  const float* A_first  = (const float*)d_in[1];
  const float* A_sub    = (const float*)d_in[2];
  // d_in[3] = coords (unused: pe_size == 0)
  const float* theta0   = (const float*)d_in[4];
  const float* bias0    = (const float*)d_in[5];
  const float* tc0_w    = (const float*)d_in[6];
  const float* tc0_b    = (const float*)d_in[7];
  const float* theta1   = (const float*)d_in[8];
  const float* bias1    = (const float*)d_in[9];
  const float* tc1_w    = (const float*)d_in[10];
  const float* tc1_b    = (const float*)d_in[11];
  const float* outc_w   = (const float*)d_in[12];
  const float* outc_b   = (const float*)d_in[13];
  const float* lin_w    = (const float*)d_in[14];
  const float* lin_b    = (const float*)d_in[15];
  float* out = (float*)d_out;

  // Workspace layout (bytes). Total ~60.1 MB.
  char* ws = (char*)d_ws;
  float*  stats0 = (float*)(ws + 0);            //  64 KB
  float*  stats1 = (float*)(ws + 65536);        //  64 KB
  __bf16* hb0    = (__bf16*)(ws + 131072);      //  64 KB  [16][N]
  __bf16* h2b0   = (__bf16*)(ws + 196608);      //  64 KB
  float*  F0     = (float*)(ws + 262144);       // 896 KB  [7][N][16]
  float*  H1     = (float*)(ws + 1179648);      // 2.25 MB [N][288]
  float*  H2     = (float*)(ws + 3538944);      // 2.25 MB
  __bf16* hb1    = (__bf16*)(ws + 5898240);     // 1.125 MB [288][N]
  __bf16* h2b1   = (__bf16*)(ws + 7077888);     // 1.125 MB
  float*  F1     = (float*)(ws + 8257536);      // 15.75 MB [7][N][288]
  float*  H3     = (float*)(ws + 24772608);     // 2.25 MB
  float*  H4     = (float*)(ws + 27131904);     // 2.25 MB
  __bf16* Wbuf   = (__bf16*)(ws + 29491200);    // 32 MB   [4][128][64][32][16]

  const int npo  = kN * kP * kHid;               // 589824
  const int blk  = 256;
  const int gNpo = (npo + blk - 1) / blk;        // 2304

  // --- per-row adjacency stats for both graphs ---
  rowstats_kernel<<<kN, 256, 0, stream>>>(A_first, stats0);
  rowstats_kernel<<<kN, 256, 0, stream>>>(A_sub,   stats1);

  // --- precompute swizzled bf16 PNA weight fragments for A_sub ---
  wprep_kernel<<<(kTB * kKT * 32 + blk - 1) / blk, blk, 0, stream>>>(A_sub, stats1, Wbuf);

  // --- layer 0: spatial conv on X (cols = 9, padded to 16), fused weights ---
  prep0_kernel<<<(kC1 * kN + blk - 1) / blk, blk, 0, stream>>>(X, hb0, h2b0);
  agg_wmma_fused_kernel<<<dim3(kTB, kC1 / 16), 32, 0, stream>>>(A_first, stats0, hb0, h2b0, F0, kC1);
  theta0_kernel<<<gNpo, blk, 0, stream>>>(F0, theta0, bias0, H1);

  // --- temporal conv0 + GLU ---
  tconv0_glu_kernel<<<gNpo, blk, 0, stream>>>(H1, tc0_w, tc0_b, H2);

  // --- layer 1: spatial conv with scalers (cols = 288), streaming weights ---
  prep1_kernel<<<(kC2 * kN + blk - 1) / blk, blk, 0, stream>>>(H2, hb1, h2b1);
  agg_wmma_pre_kernel<<<dim3(kTB, kC2 / 32), 32, 0, stream>>>(Wbuf, hb1, h2b1, F1, kC2);
  theta1_kernel<<<gNpo, blk, 0, stream>>>(F1, stats1, theta1, bias1, H3);

  // --- temporal conv1 + sigmoid ---
  tconv1_sig_kernel<<<gNpo, blk, 0, stream>>>(H3, tc1_w, tc1_b, H4);

  // --- head ---
  out_kernel<<<(kN + blk - 1) / blk, blk, 0, stream>>>(H4, outc_w, outc_b, lin_w, lin_b, out);
}
